// VDNet_6021544149246
// MI455X (gfx1250) — matrix operand order repaired and checked by reading, not compiled, not run on hardware
//
#include <hip/hip_runtime.h>

// ---------------- model dims ----------------
#define B_   16
#define T_   475
#define I_   37
#define L_   512
#define D_   768
#define H_   12
#define DH_  64
#define FF_  3072
#define V_   30522
#define TT_  2
#define VF_  2048
#define MS_  20
#define NL_  12

typedef __attribute__((ext_vector_type(16))) __bf16 bf16x16;
typedef __attribute__((ext_vector_type(8)))  __bf16 bf16x8;
typedef __attribute__((ext_vector_type(8)))  float  f32x8;

// fp32 -> bf16 round-to-nearest-even
__device__ __forceinline__ __bf16 f2bf(float f) {
    unsigned u = __builtin_bit_cast(unsigned, f);
    unsigned r = (u + 0x7FFFu + ((u >> 16) & 1u)) >> 16;
    unsigned short s = (unsigned short)r;
    return __builtin_bit_cast(__bf16, s);
}

__device__ __forceinline__ bf16x8 bzero8() {
    unsigned short z = 0;
    __bf16 zb = __builtin_bit_cast(__bf16, z);
    bf16x8 v;
#pragma unroll
    for (int i = 0; i < 8; ++i) v[i] = zb;
    return v;
}

__device__ __forceinline__ float gelu_new(float x) {
    return 0.5f * x * (1.0f + tanhf(0.7978845608028654f * (x + 0.044715f * x * x * x)));
}

// =====================================================================
// Generic WMMA bf16 GEMM:  C[z] = act( A[z] @ Bt[z]^T + bias )
//   A  : [M,K] bf16 row-major (lda = K)
//   Bt : [N,K] bf16 row-major (pre-transposed weight / K-contig)
//   C  : [M,N], optional fp32 and/or bf16 outputs
// Block: 128x128 C tile, 256 threads = 8 wave32 waves (2 x 4),
//        each wave computes 4 Mtiles x 2 Ntiles of 16x16 via
//        V_WMMA_F32_16X16X32_BF16.
// Staging: DOUBLE-BUFFERED LDS filled with GLOBAL_LOAD_ASYNC_TO_LDS_B128
//          (ASYNCcnt), so K-step k+1 streams into LDS while WMMAs run on k.
// =====================================================================
#define BM 128
#define BN 128
#define BK 32

__global__ void __launch_bounds__(256)
gemm_wmma(const __bf16* __restrict__ A, const __bf16* __restrict__ Bt,
          const float* __restrict__ bias,
          float* __restrict__ Cf, __bf16* __restrict__ Cb,
          int M, int N, int K,
          long sA, long sB, long sC, int act)
{
    __shared__ __bf16 As[2][BM * BK];
    __shared__ __bf16 Bs[2][BN * BK];

    const int z = blockIdx.z;
    A  += (long)z * sA;
    Bt += (long)z * sB;
    const long coff = (long)z * sC;

    const int m0 = blockIdx.y * BM;
    const int n0 = blockIdx.x * BN;
    const int tid  = threadIdx.x;
    const int wave = tid >> 5;
    const int lane = tid & 31;
    const int wm = wave >> 2;      // 0..1
    const int wn = wave & 3;       // 0..3
    const int lrow  = lane & 15;
    const int lhalf = lane >> 4;   // 0/1

    // async global->LDS stage of one 128x32 A tile + 128x32 Bt tile.
    // Each thread issues up to 4 global_load_async_to_lds_b128 (16B chunks).
    // Rows outside M/N are zero-filled with plain ds stores (disjoint
    // addresses from the async writes, so DScnt/ASYNCcnt never race).
    auto stage = [&](int k0, int buf) {
#pragma unroll
        for (int i = 0; i < 2; ++i) {
            int idx = tid + i * 256;          // 0..511
            int r = idx >> 2;                 // 0..127
            int c = (idx & 3) * 8;            // 0/8/16/24 (bf16 elems)
            int gm = m0 + r;
            unsigned lA = (unsigned)(size_t)&As[buf][r * BK + c];
            if (gm < M) {
                unsigned long long ga =
                    (unsigned long long)(size_t)(A + (long)gm * K + k0 + c);
                asm volatile("global_load_async_to_lds_b128 %0, %1, off"
                             :: "v"(lA), "v"(ga) : "memory");
            } else {
                *reinterpret_cast<bf16x8*>(&As[buf][r * BK + c]) = bzero8();
            }
            int gn = n0 + r;
            unsigned lB = (unsigned)(size_t)&Bs[buf][r * BK + c];
            if (gn < N) {
                unsigned long long gb =
                    (unsigned long long)(size_t)(Bt + (long)gn * K + k0 + c);
                asm volatile("global_load_async_to_lds_b128 %0, %1, off"
                             :: "v"(lB), "v"(gb) : "memory");
            } else {
                *reinterpret_cast<bf16x8*>(&Bs[buf][r * BK + c]) = bzero8();
            }
        }
    };

    const f32x8 zero8 = {0.f, 0.f, 0.f, 0.f, 0.f, 0.f, 0.f, 0.f};
    f32x8 acc[4][2];
#pragma unroll
    for (int mt = 0; mt < 4; ++mt)
#pragma unroll
        for (int nt = 0; nt < 2; ++nt) acc[mt][nt] = zero8;

    int buf = 0;
    stage(0, buf);
    asm volatile("s_wait_asynccnt 0x0" ::: "memory");
    __syncthreads();

    for (int k0 = 0; k0 < K; k0 += BK) {
        const int nbuf = buf ^ 1;
        const bool more = (k0 + BK < K);
        if (more) stage(k0 + BK, nbuf);   // prefetch next K-step (async)

        // ---- B fragments (contiguous k order, same as A: lanes<16 K=0..15, lanes>=16 K=16..31)
        bf16x16 bfr[2];
#pragma unroll
        for (int nt = 0; nt < 2; ++nt) {
            int n = (wn * 2 + nt) * 16 + lrow;
            const __bf16* bp = &Bs[buf][n * BK + lhalf * 16];
            bf16x8 lo = *reinterpret_cast<const bf16x8*>(bp);
            bf16x8 hi = *reinterpret_cast<const bf16x8*>(bp + 8);
#pragma unroll
            for (int i = 0; i < 8; ++i) { bfr[nt][i] = lo[i]; bfr[nt][8 + i] = hi[i]; }
        }
        // ---- A fragments + WMMA (A layout: lanes<16 K=0..15, lanes>=16 K=16..31)
#pragma unroll
        for (int mt = 0; mt < 4; ++mt) {
            int m = (wm * 4 + mt) * 16 + lrow;
            const __bf16* ap = &As[buf][m * BK + lhalf * 16];
            bf16x8 lo = *reinterpret_cast<const bf16x8*>(ap);
            bf16x8 hi = *reinterpret_cast<const bf16x8*>(ap + 8);
            bf16x16 afr;
#pragma unroll
            for (int i = 0; i < 8; ++i) { afr[i] = lo[i]; afr[8 + i] = hi[i]; }
#pragma unroll
            for (int nt = 0; nt < 2; ++nt) {
                acc[mt][nt] = __builtin_amdgcn_wmma_f32_16x16x32_bf16(
                    false, afr, false, bfr[nt], (short)0, acc[mt][nt], false, false);
            }
        }

        if (more) asm volatile("s_wait_asynccnt 0x0" ::: "memory");
        __syncthreads();
        buf = nbuf;
    }

    // ---- epilogue (C/D layout: n = tile_n*16 + lane&15 ; m = tile_m*16 + (lane>>4)*8 + r)
#pragma unroll
    for (int mt = 0; mt < 4; ++mt) {
#pragma unroll
        for (int nt = 0; nt < 2; ++nt) {
            int n = n0 + (wn * 2 + nt) * 16 + lrow;
            if (n >= N) continue;
            float bv = bias ? bias[n] : 0.f;
#pragma unroll
            for (int r = 0; r < 8; ++r) {
                int m = m0 + (wm * 4 + mt) * 16 + lhalf * 8 + r;
                if (m >= M) continue;
                float v = acc[mt][nt][r] + bv;
                if (act == 1)      v = gelu_new(v);
                else if (act == 2) v = tanhf(v);
                long off = coff + (long)m * N + n;
                if (Cf) Cf[off] = v;
                if (Cb) Cb[off] = f2bf(v);
            }
        }
    }
}

// =====================================================================
// weight prep: W[K,N] fp32  ->  Wt[N,K] bf16   (blockIdx.z = layer)
// =====================================================================
__global__ void __launch_bounds__(256)
transpose_cvt(const float* __restrict__ W, __bf16* __restrict__ Wt, int K, int N)
{
    long base = (long)blockIdx.z * K * N;
    long idx = (long)blockIdx.x * 256 + threadIdx.x;
    if (idx >= (long)K * N) return;
    int n = (int)(idx / K);
    int k = (int)(idx % K);
    Wt[base + idx] = f2bf(W[base + (long)k * N + n]);
}

// =====================================================================
// embeddings
// =====================================================================
__global__ void __launch_bounds__(256)
emb_txt_kernel(const int* __restrict__ txt, const int* __restrict__ spos,
               const int* __restrict__ tok,
               const float* __restrict__ wemb, const float* __restrict__ pemb,
               const float* __restrict__ temb, float* __restrict__ out)
{
    long idx = (long)blockIdx.x * 256 + threadIdx.x;
    const long total = (long)B_ * T_ * D_;
    if (idx >= total) return;
    int d = (int)(idx % D_);
    long bt = idx / D_;
    int t = (int)(bt % T_);
    int b = (int)(bt / T_);
    int w  = txt[(long)b * T_ + t];
    int tt = tok[(long)b * T_ + t];
    int sp = spos[(long)b * T_ + t];
    int j = d >> 1;
    float div = 1.f / (20.f + __expf((float)(2 * j) * (9.210340371976184f / 768.f)));
    float ang = (float)sp * div;
    float pe  = ((d & 1) == 0 ? sinf(ang) : cosf(ang)) * 0.01f;
    out[((long)b * L_ + t) * D_ + d] =
        wemb[(long)w * D_ + d] + pemb[(long)t * D_ + d] + temb[(long)tt * D_ + d] + pe;
}

__global__ void __launch_bounds__(256)
emb_img_kernel(const float* __restrict__ img, const float* __restrict__ loc,
               const float* __restrict__ imgW, const float* __restrict__ imgB,
               const float* __restrict__ locW, const float* __restrict__ locB,
               float* __restrict__ out)
{
    long idx = (long)blockIdx.x * 256 + threadIdx.x;
    const long total = (long)B_ * I_ * D_;
    if (idx >= total) return;
    int d = (int)(idx % D_);
    long bi = idx / D_;
    int i = (int)(bi % I_);
    int b = (int)(bi / I_);
    const float* xr = img + ((long)b * I_ + i) * VF_;
    float s = imgB[d] + locB[d];
    for (int f = 0; f < VF_; ++f) s += xr[f] * imgW[(long)f * D_ + d];
    const float* lr = loc + ((long)b * I_ + i) * 5;
    for (int f = 0; f < 5; ++f) s += lr[f] * locW[f * D_ + d];
    out[((long)b * L_ + T_ + i) * D_ + d] = s;
}

// =====================================================================
// LayerNorm over D=768 (+ optional residual); writes fp32 + bf16
// one 256-thread block per row, 3 elements per thread
// =====================================================================
__global__ void __launch_bounds__(256)
ln_kernel(const float* __restrict__ x, const float* __restrict__ res,
          const float* __restrict__ g, const float* __restrict__ bb,
          float* __restrict__ yf, __bf16* __restrict__ yb)
{
    __shared__ float red[256];
    long row = blockIdx.x;
    const float* xr = x + row * D_;
    const float* rr = res ? res + row * D_ : nullptr;
    int t = threadIdx.x;
    float v[3];
    float s = 0.f;
#pragma unroll
    for (int i = 0; i < 3; ++i) {
        int c = t + i * 256;
        float u = xr[c] + (rr ? rr[c] : 0.f);
        v[i] = u; s += u;
    }
    red[t] = s; __syncthreads();
    for (int st = 128; st > 0; st >>= 1) { if (t < st) red[t] += red[t + st]; __syncthreads(); }
    float mean = red[0] * (1.f / 768.f);
    __syncthreads();
    float q = 0.f;
#pragma unroll
    for (int i = 0; i < 3; ++i) { float d = v[i] - mean; q += d * d; }
    red[t] = q; __syncthreads();
    for (int st = 128; st > 0; st >>= 1) { if (t < st) red[t] += red[t + st]; __syncthreads(); }
    float rstd = rsqrtf(red[0] * (1.f / 768.f) + 1e-12f);
#pragma unroll
    for (int i = 0; i < 3; ++i) {
        int c = t + i * 256;
        float y = (v[i] - mean) * rstd * g[c] + bb[c];
        if (yf) yf[row * D_ + c] = y;
        if (yb) yb[row * D_ + c] = f2bf(y);
    }
}

// =====================================================================
// masked softmax over L=512: row = (b,h,q); writes bf16 attn
// =====================================================================
__global__ void __launch_bounds__(256)
softmax_kernel(const float* __restrict__ S, const float* __restrict__ mask,
               __bf16* __restrict__ P)
{
    __shared__ float red[256];
    long row = blockIdx.x;                // B*H*L rows
    long bh = row / L_;
    int b = (int)(bh / H_);
    const float* sr = S + row * L_;
    const float* mr = mask + (long)b * L_;
    int t = threadIdx.x;
    float v[2];
    float mx = -3.4e38f;
#pragma unroll
    for (int i = 0; i < 2; ++i) {
        int c = t + i * 256;
        float u = sr[c] * 0.125f - 1e30f * mr[c];
        v[i] = u; mx = fmaxf(mx, u);
    }
    red[t] = mx; __syncthreads();
    for (int st = 128; st > 0; st >>= 1) { if (t < st) red[t] = fmaxf(red[t], red[t + st]); __syncthreads(); }
    mx = red[0]; __syncthreads();
    float s = 0.f;
#pragma unroll
    for (int i = 0; i < 2; ++i) { v[i] = __expf(v[i] - mx); s += v[i]; }
    red[t] = s; __syncthreads();
    for (int st = 128; st > 0; st >>= 1) { if (t < st) red[t] += red[t + st]; __syncthreads(); }
    float inv = 1.f / red[0];
#pragma unroll
    for (int i = 0; i < 2; ++i) P[row * L_ + t + i * 256] = f2bf(v[i] * inv);
}

// =====================================================================
// head split / merge
// =====================================================================
__global__ void __launch_bounds__(256)
qkv_to_heads(const float* __restrict__ Q, const float* __restrict__ Kk,
             const float* __restrict__ V,
             __bf16* __restrict__ Qh, __bf16* __restrict__ Kh, __bf16* __restrict__ Vt)
{
    long idx = (long)blockIdx.x * 256 + threadIdx.x;
    const long total = (long)B_ * L_ * D_;
    if (idx >= total) return;
    int d = (int)(idx % D_);
    long bl = idx / D_;
    int l = (int)(bl % L_);
    int b = (int)(bl / L_);
    int h = d / DH_, dh = d % DH_;
    long hoff = (((long)b * H_ + h) * L_ + l) * DH_ + dh;
    Qh[hoff] = f2bf(Q[idx]);
    Kh[hoff] = f2bf(Kk[idx]);
    Vt[(((long)b * H_ + h) * DH_ + dh) * L_ + l] = f2bf(V[idx]);
}

__global__ void __launch_bounds__(256)
merge_heads(const float* __restrict__ ctx, __bf16* __restrict__ out)
{
    long idx = (long)blockIdx.x * 256 + threadIdx.x;   // output index [b,l,h*DH+dh]
    const long total = (long)B_ * L_ * D_;
    if (idx >= total) return;
    int d = (int)(idx % D_);
    long bl = idx / D_;
    int l = (int)(bl % L_);
    int b = (int)(bl / L_);
    int h = d / DH_, dh = d % DH_;
    out[idx] = f2bf(ctx[(((long)b * H_ + h) * L_ + l) * DH_ + dh]);
}

// =====================================================================
// host
// =====================================================================
extern "C" void kernel_launch(void* const* d_in, const int* in_sizes, int n_in,
                              void* d_out, int out_size, void* d_ws, size_t ws_size,
                              hipStream_t stream)
{
    (void)in_sizes; (void)n_in; (void)out_size; (void)ws_size;

    // ---- inputs (setup_inputs order; layer_params dict flattened in order)
    const int*   input_txt   = (const int*)  d_in[0];
    const int*   sentence    = (const int*)  d_in[1];
    const float* input_img   = (const float*)d_in[2];
    const float* img_loc     = (const float*)d_in[3];
    const int*   token_type  = (const int*)  d_in[4];
    const float* pad_mask    = (const float*)d_in[5];
    const float* word_emb    = (const float*)d_in[6];
    const float* pos_emb     = (const float*)d_in[7];
    const float* tok_emb     = (const float*)d_in[8];
    const float* img_W       = (const float*)d_in[9];
    const float* img_b       = (const float*)d_in[10];
    const float* loc_W       = (const float*)d_in[11];
    const float* loc_b       = (const float*)d_in[12];
    const float* emb_ln_g    = (const float*)d_in[13];
    const float* emb_ln_b    = (const float*)d_in[14];
    const float* pool_W      = (const float*)d_in[15];
    const float* pool_b      = (const float*)d_in[16];
    const float* Wq = (const float*)d_in[17]; const float* bq = (const float*)d_in[18];
    const float* Wk = (const float*)d_in[19]; const float* bk = (const float*)d_in[20];
    const float* Wv = (const float*)d_in[21]; const float* bv = (const float*)d_in[22];
    const float* Wo = (const float*)d_in[23]; const float* bo = (const float*)d_in[24];
    const float* ln1_g = (const float*)d_in[25]; const float* ln1_b = (const float*)d_in[26];
    const float* W1 = (const float*)d_in[27]; const float* b1 = (const float*)d_in[28];
    const float* W2 = (const float*)d_in[29]; const float* b2 = (const float*)d_in[30];
    const float* ln2_g = (const float*)d_in[31]; const float* ln2_b = (const float*)d_in[32];

    float* out = (float*)d_out;

    // ---- workspace carve
    const long BLD = (long)B_ * L_ * D_;            // 6,291,456
    const long DD  = (long)D_ * D_;
    const long DFF = (long)D_ * FF_;
    const long SCE = (long)B_ * H_ * L_ * L_;       // 50,331,648
    const long FFE = (long)B_ * L_ * FF_;

    char* p = (char*)d_ws;
    auto carve = [&](size_t bytes) -> void* {
        void* r = (void*)p;
        p += (bytes + 255) & ~(size_t)255;
        return r;
    };
    __bf16* wqt  = (__bf16*)carve((size_t)NL_ * DD  * 2);
    __bf16* wkt  = (__bf16*)carve((size_t)NL_ * DD  * 2);
    __bf16* wvt  = (__bf16*)carve((size_t)NL_ * DD  * 2);
    __bf16* wot  = (__bf16*)carve((size_t)NL_ * DD  * 2);
    __bf16* w1t  = (__bf16*)carve((size_t)NL_ * DFF * 2);
    __bf16* w2t  = (__bf16*)carve((size_t)NL_ * DFF * 2);
    __bf16* plt  = (__bf16*)carve((size_t)DD * 2);
    float*  hf   = (float*) carve((size_t)BLD * 4);
    __bf16* hb   = (__bf16*)carve((size_t)BLD * 2);
    float*  af   = (float*) carve((size_t)BLD * 4);
    __bf16* ab   = (__bf16*)carve((size_t)BLD * 2);
    float*  qf   = (float*) carve((size_t)BLD * 4);
    float*  kf   = (float*) carve((size_t)BLD * 4);
    float*  vf   = (float*) carve((size_t)BLD * 4);
    __bf16* qh   = (__bf16*)carve((size_t)BLD * 2);
    __bf16* kh   = (__bf16*)carve((size_t)BLD * 2);
    __bf16* vt   = (__bf16*)carve((size_t)BLD * 2);
    float*  sc   = (float*) carve((size_t)SCE * 4);
    __bf16* at   = (__bf16*)carve((size_t)SCE * 2);
    float*  cxf  = (float*) carve((size_t)BLD * 4);
    __bf16* cxb  = (__bf16*)carve((size_t)BLD * 2);
    __bf16* ffb  = (__bf16*)carve((size_t)FFE * 2);
    float*  t0   = (float*) carve((size_t)BLD * 4);

    auto gemm = [&](const __bf16* A, const __bf16* Bt, const float* bias,
                    float* Cf, __bf16* Cb, int M, int N, int K,
                    long sA, long sB, long sC, int batch, int act) {
        dim3 g((N + BN - 1) / BN, (M + BM - 1) / BM, batch);
        gemm_wmma<<<g, 256, 0, stream>>>(A, Bt, bias, Cf, Cb, M, N, K, sA, sB, sC, act);
    };
    auto tconv = [&](const float* W, __bf16* Wt, int K, int N, int layers) {
        long per = (long)K * N;
        dim3 g((unsigned)((per + 255) / 256), 1, layers);
        transpose_cvt<<<g, 256, 0, stream>>>(W, Wt, K, N);
    };

    // ---- weight prep: fp32 [K,N] -> bf16 [N,K]
    tconv(Wq, wqt, D_, D_, NL_);
    tconv(Wk, wkt, D_, D_, NL_);
    tconv(Wv, wvt, D_, D_, NL_);
    tconv(Wo, wot, D_, D_, NL_);
    tconv(W1, w1t, D_, FF_, NL_);
    tconv(W2, w2t, FF_, D_, NL_);
    tconv(pool_W, plt, D_, D_, 1);

    // ---- embeddings -> t0, then LN -> hf/hb
    {
        long n1 = (long)B_ * T_ * D_;
        emb_txt_kernel<<<(unsigned)((n1 + 255) / 256), 256, 0, stream>>>(
            input_txt, sentence, token_type, word_emb, pos_emb, tok_emb, t0);
        long n2 = (long)B_ * I_ * D_;
        emb_img_kernel<<<(unsigned)((n2 + 255) / 256), 256, 0, stream>>>(
            input_img, img_loc, img_W, img_b, loc_W, loc_b, t0);
        ln_kernel<<<(unsigned)(B_ * L_), 256, 0, stream>>>(t0, nullptr, emb_ln_g, emb_ln_b, hf, hb);
    }

    const int M  = B_ * L_;          // 8192
    const long rsh = (long)B_ * L_ * D_;

    for (int l = 0; l < NL_; ++l) {
        const __bf16* wq_l = wqt + (long)l * DD;
        const __bf16* wk_l = wkt + (long)l * DD;
        const __bf16* wv_l = wvt + (long)l * DD;
        const __bf16* wo_l = wot + (long)l * DD;
        const __bf16* w1_l = w1t + (long)l * DFF;
        const __bf16* w2_l = w2t + (long)l * DFF;

        // QKV projections
        gemm(hb, wq_l, bq + (long)l * D_, qf, nullptr, M, D_, D_, 0, 0, 0, 1, 0);
        gemm(hb, wk_l, bk + (long)l * D_, kf, nullptr, M, D_, D_, 0, 0, 0, 1, 0);
        gemm(hb, wv_l, bv + (long)l * D_, vf, nullptr, M, D_, D_, 0, 0, 0, 1, 0);
        qkv_to_heads<<<(unsigned)((rsh + 255) / 256), 256, 0, stream>>>(qf, kf, vf, qh, kh, vt);

        // scores = Qh @ Kh^T  (batched per head: 512x512x64)
        gemm(qh, kh, nullptr, sc, nullptr, L_, L_, DH_,
             (long)L_ * DH_, (long)L_ * DH_, (long)L_ * L_, B_ * H_, 0);
        softmax_kernel<<<(unsigned)(B_ * H_ * L_), 256, 0, stream>>>(sc, pad_mask, at);

        // ctx = attn @ V  (batched per head: 512x64x512)
        gemm(at, vt, nullptr, cxf, nullptr, L_, DH_, L_,
             (long)L_ * L_, (long)DH_ * L_, (long)L_ * DH_, B_ * H_, 0);
        merge_heads<<<(unsigned)((rsh + 255) / 256), 256, 0, stream>>>(cxf, cxb);

        // out projection + residual LN1
        gemm(cxb, wo_l, bo + (long)l * D_, t0, nullptr, M, D_, D_, 0, 0, 0, 1, 0);
        ln_kernel<<<(unsigned)(B_ * L_), 256, 0, stream>>>(
            t0, hf, ln1_g + (long)l * D_, ln1_b + (long)l * D_, af, ab);

        // FFN
        gemm(ab, w1_l, b1 + (long)l * FF_, nullptr, ffb, M, FF_, D_, 0, 0, 0, 1, 1);  // gelu
        gemm(ffb, w2_l, b2 + (long)l * D_, t0, nullptr, M, D_, FF_, 0, 0, 0, 1, 0);
        ln_kernel<<<(unsigned)(B_ * L_), 256, 0, stream>>>(
            t0, af, ln2_g + (long)l * D_, ln2_b + (long)l * D_, hf, hb);
    }

    // pooler: tanh(h @ pool_W + pool_b) -> d_out fp32
    gemm(hb, plt, pool_b, out, nullptr, M, D_, D_, 0, 0, 0, 1, 2);
}
